// MMOE_37091337568698
// MI455X (gfx1250) — matrix operand order repaired and checked by reading, not compiled, not run on hardware
//
#include <hip/hip_runtime.h>
#include <cstdint>
#include <cstddef>

// ---------------- problem dims ----------------
#define B_    16384
#define DIN   1024
#define NE    8
#define H1_   2048
#define H2_   2048
#define D2_   1024
#define NT    2
#define HT_   512
#define LEAK  0.01f

// ---------------- types ----------------
typedef __attribute__((ext_vector_type(16))) __bf16 v16bf;
typedef __attribute__((ext_vector_type(8)))  __bf16 v8bf;
typedef __attribute__((ext_vector_type(8)))  float  v8f;
typedef __attribute__((ext_vector_type(4)))  float  v4f;

static __device__ __forceinline__ __bf16 f2bf(float f) {
  unsigned u = __builtin_bit_cast(unsigned, f);
  u += 0x7fffu + ((u >> 16) & 1u);          // round-to-nearest-even
  unsigned short h = (unsigned short)(u >> 16);
  return __builtin_bit_cast(__bf16, h);
}
static __device__ __forceinline__ float bf2f(__bf16 h) {
  unsigned short s = __builtin_bit_cast(unsigned short, h);
  unsigned u = ((unsigned)s) << 16;
  return __builtin_bit_cast(float, u);
}

// ---------------- CDNA5 async copy: global -> LDS (ASYNCcnt path) ----------------
static __device__ __forceinline__ void async_g2l_b128(const __bf16* gsrc, __bf16* lds_dst) {
  // LDS aperture: low 32 bits of the generic pointer are the LDS byte offset.
  unsigned lofs = (unsigned)(size_t)lds_dst;
  asm volatile("global_load_async_to_lds_b128 %0, %1, off"
               :: "v"(lofs), "v"(gsrc)
               : "memory");
}
static __device__ __forceinline__ void wait_async0() {
#if __has_builtin(__builtin_amdgcn_s_wait_asynccnt)
  __builtin_amdgcn_s_wait_asynccnt(0);
#else
  asm volatile("s_wait_asynccnt 0x0" ::: "memory");
#endif
}

// ---------------- f32 -> bf16 convert (xv) ----------------
__global__ __launch_bounds__(256) void cvt4_kernel(const float* __restrict__ in,
                                                   __bf16* __restrict__ out,
                                                   long n4) {
  long i = (long)blockIdx.x * 256 + threadIdx.x;
  if (i >= n4) return;
  v4f x = *(const v4f*)(in + i * 4);
  __bf16 t[4];
#pragma unroll
  for (int j = 0; j < 4; ++j) t[j] = f2bf(x[j]);
  unsigned long long packed;
  __builtin_memcpy(&packed, t, 8);
  *(unsigned long long*)(out + i * 4) = packed;
}

// ---------------- tiled transpose + convert: f32 [R][C] -> bf16 [C][R] ----------------
__global__ __launch_bounds__(256) void tcvt_kernel(const float* __restrict__ in,
                                                   __bf16* __restrict__ out,
                                                   int R, int C) {
  __shared__ float tile[32][33];
  const size_t mo = (size_t)blockIdx.z * (size_t)R * (size_t)C;
  in += mo;
  out += mo;
  const int c0 = blockIdx.x * 32;
  const int r0 = blockIdx.y * 32;
#pragma unroll
  for (int i = threadIdx.y; i < 32; i += 8) {
    tile[i][threadIdx.x] = in[(size_t)(r0 + i) * C + (c0 + threadIdx.x)];
  }
  __syncthreads();
#pragma unroll
  for (int i = threadIdx.y; i < 32; i += 8) {
    out[(size_t)(c0 + i) * R + (r0 + threadIdx.x)] = f2bf(tile[threadIdx.x][i]);
  }
}

// ---------------- WMMA fragment loads from LDS ----------------
// A 16x32 bf16 fragment: lane l (l16=l%16,row) holds K = {8h..8h+7} U {16+8h..16+8h+7}
static __device__ __forceinline__ v16bf load_frag_a(const __bf16* base) {
  v8bf lo = *(const v8bf*)(base);        // k = 8h .. 8h+7
  v8bf hi = *(const v8bf*)(base + 16);   // k = 16+8h ..
  v16bf r;
#pragma unroll
  for (int j = 0; j < 8; ++j) { r[j] = lo[j]; r[j + 8] = hi[j]; }
  return r;
}
// B 32x16 bf16 fragment (from N-major LDS tile): lane l holds K = {8h..8h+7} U {16+8h..16+8h+7}, N = l%16
static __device__ __forceinline__ v16bf load_frag_b(const __bf16* base) {
  v8bf lo = *(const v8bf*)(base);        // k = 8h .. 8h+7
  v8bf hi = *(const v8bf*)(base + 16);   // k = 16+8h ..
  v16bf r;
#pragma unroll
  for (int j = 0; j < 8; ++j) { r[j] = lo[j]; r[j + 8] = hi[j]; }
  return r;
}

// ---------------- bf16 GEMM: C[M,N] = act(A[M,K] @ Wt[N,K]^T + bias[N]) ----------------
// ACT: 0 = relu, 1 = leaky relu, 2 = none
#define BM 128
#define BN 128
#define BK 32
#define LDSP 40   // padded LDS row length (bf16 elems); row stride 80B = 5*16B

template <int ACT>
__global__ __launch_bounds__(256) void gemm_bf16_kernel(
    const __bf16* __restrict__ A, const __bf16* __restrict__ Wt,
    const float* __restrict__ bias, __bf16* __restrict__ C,
    int M, int N, int K,
    size_t sA, size_t sW, size_t sBias, size_t sC) {
  __shared__ alignas(16) __bf16 Ash[2][BM * LDSP];
  __shared__ alignas(16) __bf16 Bsh[2][BN * LDSP];

  const int z = blockIdx.z;
  A    += (size_t)z * sA;
  Wt   += (size_t)z * sW;
  bias += (size_t)z * sBias;
  C    += (size_t)z * sC;

  const int m0 = blockIdx.y * BM;
  const int n0 = blockIdx.x * BN;
  const int tid  = threadIdx.x;
  const int wave = tid >> 5;
  const int lane = tid & 31;
  const int wm = wave >> 1;   // 0..3 -> 32-row slab
  const int wn = wave & 1;    // 0..1 -> 64-col slab
  const int half = lane >> 4; // 0/1
  const int l16  = lane & 15;

  // per-thread staging chunks: elements idx = (tid + i*256)*8, i = 0,1
  const int r0c = (tid * 8) >> 5;          // row for chunk 0
  const int c0c = (tid * 8) & 31;          // col for chunk 0 (0,8,16,24)
  const int r1c = ((tid + 256) * 8) >> 5;  // row for chunk 1
  const int c1c = ((tid + 256) * 8) & 31;

  v8f acc[2][4];
#pragma unroll
  for (int i = 0; i < 2; ++i)
#pragma unroll
    for (int j = 0; j < 4; ++j)
#pragma unroll
      for (int v = 0; v < 8; ++v) acc[i][j][v] = 0.0f;

  const int nIter = K / BK;

  // prologue: async-stage tile 0 into buffer 0
  {
    const int k0 = 0;
    async_g2l_b128(&A [(size_t)(m0 + r0c) * K + k0 + c0c], &Ash[0][r0c * LDSP + c0c]);
    async_g2l_b128(&Wt[(size_t)(n0 + r0c) * K + k0 + c0c], &Bsh[0][r0c * LDSP + c0c]);
    async_g2l_b128(&A [(size_t)(m0 + r1c) * K + k0 + c1c], &Ash[0][r1c * LDSP + c1c]);
    async_g2l_b128(&Wt[(size_t)(n0 + r1c) * K + k0 + c1c], &Bsh[0][r1c * LDSP + c1c]);
  }

  for (int it = 0; it < nIter; ++it) {
    const int buf = it & 1;
    wait_async0();       // my async writes for tile `it` have landed in LDS
    __syncthreads();     // everyone's writes landed; everyone done reading buf^1

    if (it + 1 < nIter) {
      const int k1 = (it + 1) * BK;
      __bf16* an = Ash[buf ^ 1];
      __bf16* bn = Bsh[buf ^ 1];
      async_g2l_b128(&A [(size_t)(m0 + r0c) * K + k1 + c0c], &an[r0c * LDSP + c0c]);
      async_g2l_b128(&Wt[(size_t)(n0 + r0c) * K + k1 + c0c], &bn[r0c * LDSP + c0c]);
      async_g2l_b128(&A [(size_t)(m0 + r1c) * K + k1 + c1c], &an[r1c * LDSP + c1c]);
      async_g2l_b128(&Wt[(size_t)(n0 + r1c) * K + k1 + c1c], &bn[r1c * LDSP + c1c]);
    }
    if (it + 2 < nIter) {
      const int k2 = (it + 2) * BK;   // warm L2 two tiles ahead
      __builtin_prefetch(&A [(size_t)(m0 + (tid >> 1)) * K + k2 + (tid & 1) * 16], 0, 3);
      __builtin_prefetch(&Wt[(size_t)(n0 + (tid >> 1)) * K + k2 + (tid & 1) * 16], 0, 3);
    }

    const __bf16* as = Ash[buf];
    const __bf16* bs = Bsh[buf];
    v16bf afrag[2], bfrag[4];
#pragma unroll
    for (int i = 0; i < 2; ++i)
      afrag[i] = load_frag_a(&as[(wm * 32 + i * 16 + l16) * LDSP + half * 8]);
#pragma unroll
    for (int j = 0; j < 4; ++j)
      bfrag[j] = load_frag_b(&bs[(wn * 64 + j * 16 + l16) * LDSP + half * 8]);

#pragma unroll
    for (int i = 0; i < 2; ++i)
#pragma unroll
      for (int j = 0; j < 4; ++j)
        acc[i][j] = __builtin_amdgcn_wmma_f32_16x16x32_bf16(
            false, afrag[i], false, bfrag[j], (short)0, acc[i][j], false, false);
  }

  // epilogue: bias + activation + bf16 store
#pragma unroll
  for (int j = 0; j < 4; ++j) {
    const int col = n0 + wn * 64 + j * 16 + l16;
    const float bv = bias[col];
#pragma unroll
    for (int i = 0; i < 2; ++i) {
#pragma unroll
      for (int v = 0; v < 8; ++v) {
        const int row = m0 + wm * 32 + i * 16 + v + 8 * half;
        float x = acc[i][j][v] + bv;
        if (ACT == 0) x = fmaxf(x, 0.0f);
        else if (ACT == 1) x = (x > 0.0f) ? x : LEAK * x;
        C[(size_t)row * N + col] = f2bf(x);
      }
    }
  }
}

// ---------------- gates: softmax(xv @ Wg[t] + bg[t]) over E ----------------
__global__ __launch_bounds__(256) void gates_kernel(
    const float* __restrict__ xv, const float* __restrict__ Wg,
    const float* __restrict__ bg, float* __restrict__ gates) {
  const int b = blockIdx.x * 256 + threadIdx.x;
  const int t = blockIdx.y;
  float acc[NE];
#pragma unroll
  for (int e = 0; e < NE; ++e) acc[e] = bg[t * NE + e];
  const float* wg = Wg + (size_t)t * DIN * NE;
  const float* xr = xv + (size_t)b * DIN;
  for (int i = 0; i < DIN; ++i) {
    const float x = xr[i];
    const v4f w0 = *(const v4f*)&wg[i * NE];
    const v4f w1 = *(const v4f*)&wg[i * NE + 4];
#pragma unroll
    for (int e = 0; e < 4; ++e) { acc[e] += x * w0[e]; acc[e + 4] += x * w1[e]; }
  }
  float mx = acc[0];
#pragma unroll
  for (int e = 1; e < NE; ++e) mx = fmaxf(mx, acc[e]);
  float s = 0.0f;
#pragma unroll
  for (int e = 0; e < NE; ++e) { acc[e] = __expf(acc[e] - mx); s += acc[e]; }
  const float inv = 1.0f / s;
#pragma unroll
  for (int e = 0; e < NE; ++e)
    gates[((size_t)t * B_ + b) * NE + e] = acc[e] * inv;
}

// ---------------- mixture: Tin[t,b,d] = sum_e gates[t,b,e] * Eout[e,b,d] ----------------
__global__ __launch_bounds__(256) void mix_kernel(
    const float* __restrict__ gates, const __bf16* __restrict__ Eout,
    __bf16* __restrict__ Tin) {
  const long long gid = (long long)blockIdx.x * 256 + threadIdx.x;
  const int d  = (int)(gid % D2_);
  const int bb = (int)((gid / D2_) % B_);
  const int t  = (int)(gid / ((long long)D2_ * B_));
  const float* g = &gates[((size_t)t * B_ + bb) * NE];
  float s = 0.0f;
#pragma unroll
  for (int e = 0; e < NE; ++e)
    s += g[e] * bf2f(Eout[((size_t)e * B_ + bb) * D2_ + d]);
  Tin[gid] = f2bf(s);
}

// ---------------- tower head: sigmoid(Th[t,b,:] . Wt2[t] + bt2[t]) ----------------
__global__ __launch_bounds__(256) void tower2_kernel(
    const __bf16* __restrict__ Th, const float* __restrict__ Wt2,
    const float* __restrict__ bt2, float* __restrict__ out) {
  const int b = blockIdx.x * 256 + threadIdx.x;
  const int t = blockIdx.y;
  const __bf16* hr = &Th[((size_t)t * B_ + b) * HT_];
  const float*  wr = &Wt2[(size_t)t * HT_];
  float s = bt2[t];
  for (int h = 0; h < HT_; h += 4) {
#pragma unroll
    for (int j = 0; j < 4; ++j) s += bf2f(hr[h + j]) * wr[h + j];
  }
  out[(size_t)t * B_ + b] = 1.0f / (1.0f + __expf(-s));
}

// ---------------- launch ----------------
extern "C" void kernel_launch(void* const* d_in, const int* in_sizes, int n_in,
                              void* d_out, int out_size, void* d_ws, size_t ws_size,
                              hipStream_t stream) {
  const float* xv  = (const float*)d_in[0];
  const float* W1  = (const float*)d_in[1];
  const float* b1  = (const float*)d_in[2];
  const float* W2  = (const float*)d_in[3];
  const float* b2  = (const float*)d_in[4];
  const float* W3  = (const float*)d_in[5];
  const float* b3  = (const float*)d_in[6];
  const float* Wg  = (const float*)d_in[7];
  const float* bg  = (const float*)d_in[8];
  const float* Wt1 = (const float*)d_in[9];
  const float* bt1 = (const float*)d_in[10];
  const float* Wt2 = (const float*)d_in[11];
  const float* bt2 = (const float*)d_in[12];
  float* out = (float*)d_out;

  char* p = (char*)d_ws;
  __bf16* Xb   = (__bf16*)p; p += (size_t)B_ * DIN * 2;
  __bf16* W1t  = (__bf16*)p; p += (size_t)NE * H1_ * DIN * 2;
  __bf16* W2t  = (__bf16*)p; p += (size_t)NE * H2_ * H1_ * 2;
  __bf16* W3t  = (__bf16*)p; p += (size_t)NE * D2_ * H2_ * 2;
  __bf16* Wt1t = (__bf16*)p; p += (size_t)NT * HT_ * D2_ * 2;
  float*  gatesP = (float*)p; p += (size_t)NT * B_ * NE * 4;
  __bf16* bufA = (__bf16*)p; p += (size_t)NE * B_ * H1_ * 2;   // H1 acts, then Eout
  __bf16* bufB = (__bf16*)p; p += (size_t)NE * B_ * H2_ * 2;   // H2 acts, then Tin
  __bf16* Th   = (__bf16*)p; p += (size_t)NT * B_ * HT_ * 2;   // tower hidden

  // 1) convert xv -> bf16
  {
    long n4 = (long)B_ * DIN / 4;
    cvt4_kernel<<<dim3((unsigned)(n4 / 256)), dim3(256), 0, stream>>>(xv, Xb, n4);
  }
  // 2) transpose-convert weights to N-major bf16
  tcvt_kernel<<<dim3(H1_ / 32, DIN / 32, NE), dim3(32, 8), 0, stream>>>(W1, W1t, DIN, H1_);
  tcvt_kernel<<<dim3(H2_ / 32, H1_ / 32, NE), dim3(32, 8), 0, stream>>>(W2, W2t, H1_, H2_);
  tcvt_kernel<<<dim3(D2_ / 32, H2_ / 32, NE), dim3(32, 8), 0, stream>>>(W3, W3t, H2_, D2_);
  tcvt_kernel<<<dim3(HT_ / 32, D2_ / 32, NT), dim3(32, 8), 0, stream>>>(Wt1, Wt1t, D2_, HT_);

  // 3) expert layer 1: relu(Xb @ W1 + b1) -> bufA   (A batch-stride 0: shared across experts)
  gemm_bf16_kernel<0><<<dim3(H1_ / BN, B_ / BM, NE), dim3(256), 0, stream>>>(
      Xb, W1t, b1, bufA, B_, H1_, DIN,
      (size_t)0, (size_t)H1_ * DIN, (size_t)H1_, (size_t)B_ * H1_);

  // 4) expert layer 2: leaky_relu(bufA @ W2 + b2) -> bufB
  gemm_bf16_kernel<1><<<dim3(H2_ / BN, B_ / BM, NE), dim3(256), 0, stream>>>(
      bufA, W2t, b2, bufB, B_, H2_, H1_,
      (size_t)B_ * H1_, (size_t)H2_ * H1_, (size_t)H2_, (size_t)B_ * H2_);

  // 5) expert layer 3: relu(bufB @ W3 + b3) -> Eout (reuse bufA)
  __bf16* Eout = bufA;
  gemm_bf16_kernel<0><<<dim3(D2_ / BN, B_ / BM, NE), dim3(256), 0, stream>>>(
      bufB, W3t, b3, Eout, B_, D2_, H2_,
      (size_t)B_ * H2_, (size_t)D2_ * H2_, (size_t)D2_, (size_t)B_ * D2_);

  // 6) gates
  gates_kernel<<<dim3(B_ / 256, NT), dim3(256), 0, stream>>>(xv, Wg, bg, gatesP);

  // 7) mixture -> Tin (reuse bufB)
  __bf16* Tin = bufB;
  {
    long long n = (long long)NT * B_ * D2_;
    mix_kernel<<<dim3((unsigned)(n / 256)), dim3(256), 0, stream>>>(gatesP, Eout, Tin);
  }

  // 8) tower layer 1: relu(Tin @ Wt1 + bt1) -> Th
  gemm_bf16_kernel<0><<<dim3(HT_ / BN, B_ / BM, NT), dim3(256), 0, stream>>>(
      Tin, Wt1t, bt1, Th, B_, HT_, D2_,
      (size_t)B_ * D2_, (size_t)HT_ * D2_, (size_t)HT_, (size_t)B_ * HT_);

  // 9) tower head + sigmoid -> d_out ([task0 B floats][task1 B floats])
  tower2_kernel<<<dim3(B_ / 256, NT), dim3(256), 0, stream>>>(Th, Wt2, bt2, out);

  (void)in_sizes; (void)n_in; (void)out_size; (void)ws_size;
}
